// HGT_69647189671982
// MI455X (gfx1250) — matrix operand order repaired and checked by reading, not compiled, not run on hardware
//
#include <hip/hip_runtime.h>
#include <hip/hip_bf16.h>
#include <math.h>

#define NN   50000
#define EE   800000
#define DIM  128
#define HH   8
#define DKK  16
#define NTILES (NN / 16)   // 3125 exactly

typedef _Float16 v16h __attribute__((ext_vector_type(16)));
typedef _Float16 h8   __attribute__((ext_vector_type(8)));
typedef float    v8f  __attribute__((ext_vector_type(8)));
typedef float    f4   __attribute__((ext_vector_type(4)));

// ---------- helpers ----------

// order-preserving float->uint encoding for atomicMax-based segment max
__device__ inline unsigned fenc(float x) {
    unsigned b = __float_as_uint(x);
    return (b & 0x80000000u) ? ~b : (b | 0x80000000u);
}
__device__ inline float fdec(unsigned u) {
    return __uint_as_float((u & 0x80000000u) ? (u & 0x7FFFFFFFu) : ~u);
}

// pack W[k][o] (row-major 128x128) into WMMA B-operand lane order:
// B 32x16 f16 tile: lane l (lh=l/16, n=l%16) holds halves j<8 = B[lh*8+j][n], j>=8 = B[16+lh*8+j-8][n]
__device__ inline void packB(_Float16* p, int k, int o, float val) {
    int kb = k >> 5, kr = k & 31;
    int lh = (kr >> 3) & 1, j = (kr & 7) | ((kr >> 4) << 3);
    int ot = o >> 4,  n = o & 15;
    int lane = lh * 16 + n;
    p[(((kb * 8 + ot) * 32 + lane) << 4) + j] = (_Float16)val;
}

// load one 128-K B panel (4 x 32-K WMMA operands) from packed layout
__device__ inline void load_panelB(const _Float16* __restrict__ pW, int ot, int lane,
                                   v16h* B) {
#pragma unroll
    for (int kb = 0; kb < 4; ++kb) {
        const h8* bp = (const h8*)(pW + (((kb * 8 + ot) * 32 + lane) << 4));
        h8 lo = bp[0], hi = bp[1];
#pragma unroll
        for (int j = 0; j < 8; ++j) { B[kb][j] = lo[j]; B[kb][8 + j] = hi[j]; }
    }
}

// load the 4 A operands (16x128 row panel) for one node tile, f16 source
__device__ inline void load_panelA_h16(const _Float16* __restrict__ arow, int lh,
                                       v16h* A) {
#pragma unroll
    for (int kb = 0; kb < 4; ++kb) {
        h8 alo = *(const h8*)(arow + kb * 32 + lh * 8);
        h8 ahi = *(const h8*)(arow + kb * 32 + 16 + lh * 8);
#pragma unroll
        for (int j = 0; j < 8; ++j) { A[kb][j] = alo[j]; A[kb][8 + j] = ahi[j]; }
    }
}

// same but f32 source converted inline
__device__ inline void load_panelA_f32(const float* __restrict__ arow, int lh,
                                       v16h* A) {
#pragma unroll
    for (int kb = 0; kb < 4; ++kb) {
        f4 a0 = *(const f4*)(arow + kb * 32 + lh * 8);
        f4 a1 = *(const f4*)(arow + kb * 32 + lh * 8 + 4);
        f4 a2 = *(const f4*)(arow + kb * 32 + 16 + lh * 8);
        f4 a3 = *(const f4*)(arow + kb * 32 + 16 + lh * 8 + 4);
#pragma unroll
        for (int j = 0; j < 4; ++j) {
            A[kb][j]      = (_Float16)a0[j];  A[kb][4 + j]  = (_Float16)a1[j];
            A[kb][8 + j]  = (_Float16)a2[j];  A[kb][12 + j] = (_Float16)a3[j];
        }
    }
}

__device__ inline void store_tile(float* __restrict__ yrow, const v8f& acc, float bval,
                                  int lh) {
#pragma unroll
    for (int r = 0; r < 8; ++r) yrow[(size_t)(r + lh * 8) * DIM] = acc[r] + bval;
}

// ---------- kernel 0: fuse weight chains + pack all weights to f16 B-layout ----------
__global__ void hgt_fuse_pack(const float* __restrict__ Wq, const float* __restrict__ Wk,
                              const float* __restrict__ Wv, const float* __restrict__ Wmsg,
                              const float* __restrict__ Wattn, const float* __restrict__ Wa,
                              const float* __restrict__ bk, const float* __restrict__ battn,
                              const float* __restrict__ bv, const float* __restrict__ bmsg,
                              _Float16* pWq, _Float16* pWkk, _Float16* pWvv, _Float16* pWa,
                              float* bkk, float* bvv) {
    int tid = blockIdx.x * blockDim.x + threadIdx.x;
    if (tid >= DIM * DIM) return;
    int k = tid >> 7, o = tid & 127;
    float skk = 0.f, svv = 0.f;
    for (int j = 0; j < DIM; ++j) {
        skk += Wk[k * DIM + j] * Wattn[j * DIM + o];
        svv += Wv[k * DIM + j] * Wmsg[j * DIM + o];
    }
    packB(pWq,  k, o, Wq[k * DIM + o]);
    packB(pWkk, k, o, skk);
    packB(pWvv, k, o, svv);
    packB(pWa,  k, o, Wa[k * DIM + o]);
    if (k == 0) {
        float a = battn[o], b = bmsg[o];
        for (int j = 0; j < DIM; ++j) {
            a += bk[j] * Wattn[j * DIM + o];
            b += bv[j] * Wmsg[j * DIM + o];
        }
        bkk[o] = a; bvv[o] = b;
    }
}

// ---------- kernel 1: h (f32) -> f16 ----------
__global__ void hgt_cvt_h(const float* __restrict__ h, _Float16* __restrict__ h16, int n) {
    int i = blockIdx.x * blockDim.x + threadIdx.x;
    if (i < n) h16[i] = (_Float16)h[i];
}

// ---------- kernel 2: fused q/k/v projections (WMMA, 3 independent acc chains) ----------
// one wave per 16-col output tile; each A-tile load feeds 3 WMMA chains (q,k,v),
// so the f16 WMMA RAW-hazard slots are filled with independent WMMAs, and h16 is
// read once instead of three times.
__global__ __launch_bounds__(32) void hgt_gemm_qkv(const _Float16* __restrict__ X,
                                                   const _Float16* __restrict__ pWq,
                                                   const _Float16* __restrict__ pWk,
                                                   const _Float16* __restrict__ pWv,
                                                   const float* __restrict__ bq,
                                                   const float* __restrict__ bk,
                                                   const float* __restrict__ bv,
                                                   float* __restrict__ Q,
                                                   float* __restrict__ K,
                                                   float* __restrict__ V, int ntiles) {
    const int ot   = blockIdx.y;
    const int lane = threadIdx.x;
    const int lh = lane >> 4, ln = lane & 15;

    v16h Bq[4], Bk[4], Bv[4];
    load_panelB(pWq, ot, lane, Bq);
    load_panelB(pWk, ot, lane, Bk);
    load_panelB(pWv, ot, lane, Bv);
    const float bqv = bq[ot * 16 + ln];
    const float bkv = bk[ot * 16 + ln];
    const float bvv = bv[ot * 16 + ln];

    for (int tile = blockIdx.x; tile < ntiles; tile += gridDim.x) {
        const _Float16* arow = X + (size_t)(tile * 16 + ln) * DIM;
        v16h A[4];
        load_panelA_h16(arow, lh, A);
        v8f aq = {}, ak = {}, av = {};
#pragma unroll
        for (int kb = 0; kb < 4; ++kb) {
            aq = __builtin_amdgcn_wmma_f32_16x16x32_f16(false, A[kb], false, Bq[kb],
                                                        (short)0, aq, false, false);
            ak = __builtin_amdgcn_wmma_f32_16x16x32_f16(false, A[kb], false, Bk[kb],
                                                        (short)0, ak, false, false);
            av = __builtin_amdgcn_wmma_f32_16x16x32_f16(false, A[kb], false, Bv[kb],
                                                        (short)0, av, false, false);
        }
        size_t ybase = (size_t)(tile * 16) * DIM + ot * 16 + ln;
        store_tile(Q + ybase, aq, bqv, lh);
        store_tile(K + ybase, ak, bkv, lh);
        store_tile(V + ybase, av, bvv, lh);
    }
}

// ---------- kernel 2b: h_out = agg(f32) @ Wa + ba; two node tiles per iteration ----------
// two independent accumulator chains per wave hide the WMMA->WMMA hazard; the
// tile1 guard is wave-uniform so EXEC stays all-ones (WMMA requirement).
__global__ __launch_bounds__(32) void hgt_gemm_f32src(const float* __restrict__ X,
                                                      const _Float16* __restrict__ pW,
                                                      const float* __restrict__ bias,
                                                      float* __restrict__ Y, int ntiles) {
    const int ot   = blockIdx.y;
    const int lane = threadIdx.x;
    const int lh = lane >> 4, ln = lane & 15;

    v16h B[4];
    load_panelB(pW, ot, lane, B);
    const float bval = bias[ot * 16 + ln];

    for (int t0 = blockIdx.x * 2; t0 < ntiles; t0 += gridDim.x * 2) {
        const int t1 = t0 + 1;
        const bool has1 = (t1 < ntiles);
        const int t1c = has1 ? t1 : t0;   // compute garbage on the odd tail, skip store

        const float* arow0 = X + (size_t)(t0 * 16 + ln) * DIM;
        const float* arow1 = X + (size_t)(t1c * 16 + ln) * DIM;
        v16h A0[4], A1[4];
        load_panelA_f32(arow0, lh, A0);
        load_panelA_f32(arow1, lh, A1);
        v8f c0 = {}, c1 = {};
#pragma unroll
        for (int kb = 0; kb < 4; ++kb) {
            c0 = __builtin_amdgcn_wmma_f32_16x16x32_f16(false, A0[kb], false, B[kb],
                                                        (short)0, c0, false, false);
            c1 = __builtin_amdgcn_wmma_f32_16x16x32_f16(false, A1[kb], false, B[kb],
                                                        (short)0, c1, false, false);
        }
        store_tile(Y + (size_t)(t0 * 16) * DIM + ot * 16 + ln, c0, bval, lh);
        if (has1)
            store_tile(Y + (size_t)(t1 * 16) * DIM + ot * 16 + ln, c1, bval, lh);
    }
}

// ---------- kernel 3: SDDMM per (edge, head) + segmented max via ordered atomicMax ----------
__global__ void hgt_sddmm(const float* __restrict__ kbuf, const float* __restrict__ qbuf,
                          const int* __restrict__ src, const int* __restrict__ dst,
                          float* __restrict__ t, unsigned* __restrict__ menc) {
    int tid = blockIdx.x * blockDim.x + threadIdx.x;
    if (tid >= EE * HH) return;
    int e = tid >> 3, head = tid & 7;
    const float* kr = kbuf + (size_t)src[e] * DIM + head * DKK;
    const float* qr = qbuf + (size_t)dst[e] * DIM + head * DKK;
    float acc = 0.f;
#pragma unroll
    for (int d = 0; d < DKK; d += 4) {
        f4 a = *(const f4*)(kr + d);
        f4 b = *(const f4*)(qr + d);
        acc += a[0] * b[0] + a[1] * b[1] + a[2] * b[2] + a[3] * b[3];
    }
    float tv = acc * 0.25f;               // 1/sqrt(16)
    t[tid] = tv;
    atomicMax(menc + (size_t)dst[e] * HH + head, fenc(tv));
}

// ---------- kernel 4: e = exp(t - m[dst]); segmented sum ----------
__global__ void hgt_expsum(const int* __restrict__ dst, float* __restrict__ t,
                           const unsigned* __restrict__ menc, float* __restrict__ sbuf) {
    int tid = blockIdx.x * blockDim.x + threadIdx.x;
    if (tid >= EE * HH) return;
    int e = tid >> 3, head = tid & 7;
    size_t nh = (size_t)dst[e] * HH + head;
    float ev = __expf(t[tid] - fdec(menc[nh]));
    t[tid] = ev;
    atomicAdd(sbuf + nh, ev);
}

// ---------- kernel 5: normalize + SPMM scatter (agg[dst] += v[src]*attn) ----------
__global__ void hgt_spmm(const int* __restrict__ src, const int* __restrict__ dst,
                         const float* __restrict__ t, const float* __restrict__ sbuf,
                         const float* __restrict__ vbuf, float* __restrict__ agg) {
    int tid = blockIdx.x * blockDim.x + threadIdx.x;
    if (tid >= EE * HH) return;
    int e = tid >> 3, head = tid & 7;
    float attn = t[tid] / sbuf[(size_t)dst[e] * HH + head];
    const float* vr = vbuf + (size_t)src[e] * DIM + head * DKK;
    float* ar = agg + (size_t)dst[e] * DIM + head * DKK;
#pragma unroll
    for (int d = 0; d < DKK; ++d) atomicAdd(ar + d, vr[d] * attn);
}

// ---------- kernel 6: per-edge 128-dot scores ----------
__global__ void hgt_scores(const float* __restrict__ ho, const int* __restrict__ a,
                           const int* __restrict__ b, float* __restrict__ out) {
    int e = blockIdx.x * blockDim.x + threadIdx.x;
    if (e >= EE) return;
    const float* x = ho + (size_t)a[e] * DIM;
    const float* y = ho + (size_t)b[e] * DIM;
    float acc = 0.f;
#pragma unroll 8
    for (int d = 0; d < DIM; d += 4) {
        f4 xa = *(const f4*)(x + d);
        f4 yb = *(const f4*)(y + d);
        acc += xa[0] * yb[0] + xa[1] * yb[1] + xa[2] * yb[2] + xa[3] * yb[3];
    }
    out[e] = acc;
}

extern "C" void kernel_launch(void* const* d_in, const int* in_sizes, int n_in,
                              void* d_out, int out_size, void* d_ws, size_t ws_size,
                              hipStream_t stream) {
    const float* h     = (const float*)d_in[0];
    const float* Wq    = (const float*)d_in[1];
    const float* bq    = (const float*)d_in[2];
    const float* Wk    = (const float*)d_in[3];
    const float* bk    = (const float*)d_in[4];
    const float* Wv    = (const float*)d_in[5];
    const float* bv    = (const float*)d_in[6];
    const float* Wmsg  = (const float*)d_in[7];
    const float* bmsg  = (const float*)d_in[8];
    const float* Wattn = (const float*)d_in[9];
    const float* battn = (const float*)d_in[10];
    const float* Wa    = (const float*)d_in[11];
    const float* ba    = (const float*)d_in[12];
    const int* src     = (const int*)d_in[13];
    const int* dst     = (const int*)d_in[14];
    const int* nsrc    = (const int*)d_in[15];
    const int* ndst    = (const int*)d_in[16];

    char* ws = (char*)d_ws;
    size_t off = 0;
    auto carve = [&](size_t bytes) { char* p = ws + off; off += (bytes + 255) & ~(size_t)255; return p; };

    _Float16* h16  = (_Float16*)carve((size_t)NN * DIM * 2);
    _Float16* pWq  = (_Float16*)carve((size_t)DIM * DIM * 2);
    _Float16* pWkk = (_Float16*)carve((size_t)DIM * DIM * 2);
    _Float16* pWvv = (_Float16*)carve((size_t)DIM * DIM * 2);
    _Float16* pWa  = (_Float16*)carve((size_t)DIM * DIM * 2);
    float*    bkk  = (float*)carve(DIM * 4);
    float*    bvv  = (float*)carve(DIM * 4);
    float*    qbuf = (float*)carve((size_t)NN * DIM * 4);
    float*    kbuf = (float*)carve((size_t)NN * DIM * 4);
    float*    vbuf = (float*)carve((size_t)NN * DIM * 4);
    float*    tbuf = (float*)carve((size_t)EE * HH * 4);
    unsigned* menc = (unsigned*)carve((size_t)NN * HH * 4);
    float*    sbuf = (float*)carve((size_t)NN * HH * 4);
    float*    agg  = (float*)carve((size_t)NN * DIM * 4);

    float* h_out     = (float*)d_out;                  // [N,128]
    float* score_g   = h_out + (size_t)NN * DIM;       // [E]
    float* score_neg = score_g + EE;                   // [E]

    // capture-safe zero-init of accumulators (menc==0 decodes below any real score)
    hipMemsetAsync(menc, 0, (size_t)NN * HH * 4, stream);
    hipMemsetAsync(sbuf, 0, (size_t)NN * HH * 4, stream);
    hipMemsetAsync(agg,  0, (size_t)NN * DIM * 4, stream);

    // 0) fuse + pack weights
    hgt_fuse_pack<<<64, 256, 0, stream>>>(Wq, Wk, Wv, Wmsg, Wattn, Wa,
                                          bk, battn, bv, bmsg,
                                          pWq, pWkk, pWvv, pWa, bkk, bvv);
    // 1) h -> f16
    hgt_cvt_h<<<(NN * DIM + 255) / 256, 256, 0, stream>>>(h, h16, NN * DIM);

    // 2) fused node projections (WMMA): q, k-fused, v-fused in one pass
    hgt_gemm_qkv<<<dim3(625, 8, 1), 32, 0, stream>>>(h16, pWq, pWkk, pWvv,
                                                     bq, bkk, bvv,
                                                     qbuf, kbuf, vbuf, NTILES);

    // 3-5) edge softmax + SPMM
    const int eht = (EE * HH + 255) / 256;
    hgt_sddmm <<<eht, 256, 0, stream>>>(kbuf, qbuf, src, dst, tbuf, menc);
    hgt_expsum<<<eht, 256, 0, stream>>>(dst, tbuf, menc, sbuf);
    hgt_spmm  <<<eht, 256, 0, stream>>>(src, dst, tbuf, sbuf, vbuf, agg);

    // 6) h_out = agg @ Wa + ba (WMMA, f32 source converted inline, tile-paired)
    hgt_gemm_f32src<<<dim3(313, 8, 1), 32, 0, stream>>>(agg, pWa, ba, h_out, NTILES);

    // 7) edge scores
    hgt_scores<<<(EE + 255) / 256, 256, 0, stream>>>(h_out, src,  dst,  score_g);
    hgt_scores<<<(EE + 255) / 256, 256, 0, stream>>>(h_out, nsrc, ndst, score_neg);
}
